// Attention_68255620268519
// MI455X (gfx1250) — matrix operand order repaired and checked by reading, not compiled, not run on hardware
//
#include <hip/hip_runtime.h>
#include <cstdint>
#include <cstddef>

// ---------------------------------------------------------------------------
// CDNA5 (gfx1250) WMMA multi-head attention forward.
// Pipeline: [f32->f16 convert/transpose] -> [QKV GEMM (WMMA) + scatter]
//           -> [flash attention (WMMA QK^T, online softmax, WMMA PV)]
//           -> [proj GEMM (WMMA) + bias -> f32 out]
// Double-buffered LDS staging driven by gfx1250 async LDS-DMA
// (GLOBAL_LOAD_ASYNC_TO_LDS_B128 / s_wait_asynccnt), one barrier per K-step.
// ---------------------------------------------------------------------------

typedef __attribute__((ext_vector_type(16))) _Float16 v16h;
typedef __attribute__((ext_vector_type(8)))  float    v8f;
typedef __attribute__((ext_vector_type(4)))  int      v4i;

#define WMMA_F16F32(a, b, c) \
  __builtin_amdgcn_wmma_f32_16x16x32_f16(false, (a), false, (b), (short)0, (c), false, false)

#if defined(__gfx1250__) && __has_builtin(__builtin_amdgcn_global_load_async_to_lds_b128) && \
    __has_builtin(__builtin_amdgcn_s_wait_asynccnt)
#define HAVE_ASYNC_LDS 1
#else
#define HAVE_ASYNC_LDS 0
#endif

static constexpr int kB    = 16;
static constexpr int kN    = 1024;
static constexpr int kD    = 768;
static constexpr int kH    = 12;
static constexpr int kHd   = 64;
static constexpr int kM    = kB * kN;     // 16384 rows
static constexpr int kNQKV = 3 * kD;      // 2304

__device__ __forceinline__ int lane_id() { return (int)(threadIdx.x & 31u); }

// Copy 16 bytes global -> LDS. Async DMA (ASYNCcnt) when available, else sync.
__device__ __forceinline__ void copy_b128_to_lds(const _Float16* g, _Float16* l) {
#if HAVE_ASYNC_LDS
  __builtin_amdgcn_global_load_async_to_lds_b128(
      (__attribute__((address_space(1))) v4i*)g,
      (__attribute__((address_space(3))) v4i*)l, 0, 0);
#else
  *reinterpret_cast<float4*>(l) = *reinterpret_cast<const float4*>(g);
#endif
}

__device__ __forceinline__ void wait_async_lds() {
#if HAVE_ASYNC_LDS
  __builtin_amdgcn_s_wait_asynccnt(0);
#endif
}

// --- A fragment: 16x32 f16 tile, row-major in LDS (leading dim ld). ---------
// Contiguous k order used here (matches load_frag_b): lanes 0-15 (row = lane) hold K 0..15;
//             lanes 16-31 (row = lane-16) hold K 16..31.
__device__ __forceinline__ v16h load_frag_a(const _Float16* base, int ld) {
  const int lane = lane_id();
  const int row  = lane & 15;
  const int koff = (lane < 16) ? 0 : 16;
  const _Float16* p = base + row * ld + koff;
  v16h a;
#pragma unroll
  for (int i = 0; i < 8; ++i) a[i] = p[i];
#pragma unroll
  for (int i = 0; i < 8; ++i) a[8 + i] = p[8 + i];
  return a;
}

// --- B fragment: 32x16 f16 tile, stored TRANSPOSED in LDS: base[n*ld + k]. --
// Contiguous k order used here (matches load_frag_a): lanes 0-15 hold column N=lane, K=0..15;
//             lanes 16-31 hold column N=lane-16, K=16..31.
__device__ __forceinline__ v16h load_frag_b(const _Float16* base, int ld) {
  const int lane = lane_id();
  const int n    = lane & 15;
  const int koff = (lane < 16) ? 0 : 16;
  const _Float16* p = base + n * ld + koff;
  v16h b;
#pragma unroll
  for (int i = 0; i < 16; ++i) b[i] = p[i];
  return b;
}

// ---------------------------------------------------------------------------
// Kernel 0a: f32 -> f16 elementwise convert.
// ---------------------------------------------------------------------------
__global__ void cvt_f32_to_f16(const float* __restrict__ src,
                               _Float16* __restrict__ dst, size_t n) {
  size_t i = (size_t)blockIdx.x * blockDim.x + threadIdx.x;
  size_t stride = (size_t)gridDim.x * blockDim.x;
  for (; i < n; i += stride) dst[i] = (_Float16)src[i];
}

// ---------------------------------------------------------------------------
// Kernel 0b: f32 [rows][cols] -> f16 [cols][rows]  (weight transpose so GEMM
// B-tiles stage contiguously as [n][k], matching the B-fragment layout).
// ---------------------------------------------------------------------------
__global__ void cvt_transpose_f32_to_f16(const float* __restrict__ src,
                                         _Float16* __restrict__ dst,
                                         int rows, int cols) {
  size_t total = (size_t)rows * cols;
  size_t i = (size_t)blockIdx.x * blockDim.x + threadIdx.x;
  size_t stride = (size_t)gridDim.x * blockDim.x;
  for (; i < total; i += stride) {
    int r = (int)(i / cols);
    int c = (int)(i % cols);
    dst[(size_t)c * rows + r] = (_Float16)src[i];
  }
}

// ---------------------------------------------------------------------------
// WMMA GEMM:  C[M, Nout] = A[M, 768] * Bt[Nout, 768]^T  (+ bias epilogue)
// Block tile 128x128, K-step 32, double-buffered async LDS staging.
// 8 waves: wave = (wm in 0..3) x (wn in 0..1), each wave 32x64 = 2x4 tiles.
// MODE 0: scatter into q/k/v f16 [B,H,N,Hd], scale folded into Q.
// MODE 1: f32 + bias straight to output.
// ---------------------------------------------------------------------------
template <int MODE>
__global__ void gemm_wmma(const _Float16* __restrict__ A,
                          const _Float16* __restrict__ Bt,
                          const float* __restrict__ bias,
                          _Float16* __restrict__ qb,
                          _Float16* __restrict__ kb,
                          _Float16* __restrict__ vb,
                          float scale,
                          float* __restrict__ fout) {
  __shared__ __align__(16) _Float16 As[2][128 * 32];
  __shared__ __align__(16) _Float16 Bs[2][128 * 32];

  const int tid  = threadIdx.x;
  const int w    = tid >> 5;
  const int lane = tid & 31;
  const int wm   = w & 3;   // 0..3 -> rows wm*32
  const int wn   = w >> 2;  // 0..1 -> cols wn*64
  const int m0   = blockIdx.y * 128;
  const int n0   = blockIdx.x * 128;
  constexpr int KT = kD / 32;   // 24 K-steps

  v8f zero = {};
  v8f acc[2][4];
#pragma unroll
  for (int mi = 0; mi < 2; ++mi)
#pragma unroll
    for (int ni = 0; ni < 4; ++ni) acc[mi][ni] = zero;

  const int r  = tid >> 1;          // 0..127 staging row
  const int kp = (tid & 1) * 16;    // 0 or 16
  const _Float16* gA = A  + (size_t)(m0 + r) * kD + kp;
  const _Float16* gB = Bt + (size_t)(n0 + r) * kD + kp;

  auto stage = [&](int buf, int k0) {
    const _Float16* ga = gA + k0;
    const _Float16* gb = gB + k0;
    _Float16* la = &As[buf][r * 32 + kp];
    _Float16* lb = &Bs[buf][r * 32 + kp];
    copy_b128_to_lds(ga,     la);
    copy_b128_to_lds(ga + 8, la + 8);
    copy_b128_to_lds(gb,     lb);
    copy_b128_to_lds(gb + 8, lb + 8);
  };

  stage(0, 0);  // prologue

  for (int kt = 0; kt < KT; ++kt) {
    const int cur = kt & 1;
    wait_async_lds();     // drain this wave's in-flight DMA for buf[cur]
    __syncthreads();      // all waves' staging for buf[cur] visible
    if (kt + 1 < KT) stage(cur ^ 1, (kt + 1) * 32);  // overlap with compute

    v16h afr[2];
    afr[0] = load_frag_a(&As[cur][(wm * 32 + 0)  * 32], 32);
    afr[1] = load_frag_a(&As[cur][(wm * 32 + 16) * 32], 32);
#pragma unroll
    for (int ni = 0; ni < 4; ++ni) {
      v16h bfr = load_frag_b(&Bs[cur][(wn * 64 + ni * 16) * 32], 32);
      acc[0][ni] = WMMA_F16F32(afr[0], bfr, acc[0][ni]);
      acc[1][ni] = WMMA_F16F32(afr[1], bfr, acc[1][ni]);
    }
  }

  // Epilogue (C layout: lane<16 -> rows mb+0..7 col lane; lane>=16 -> rows 8..15).
  const int nl = lane & 15;
  const int mb = (lane < 16) ? 0 : 8;
#pragma unroll
  for (int mi = 0; mi < 2; ++mi) {
#pragma unroll
    for (int ni = 0; ni < 4; ++ni) {
#pragma unroll
      for (int e = 0; e < 8; ++e) {
        int grow = m0 + wm * 32 + mi * 16 + mb + e;
        int gcol = n0 + wn * 64 + ni * 16 + nl;
        float val = acc[mi][ni][e] + bias[gcol];
        if (MODE == 0) {
          int s   = gcol / kD;          // 0=q 1=k 2=v
          int rem = gcol - s * kD;
          int h   = rem >> 6;           // /64
          int d   = rem & 63;
          int b   = grow >> 10;         // /1024
          int n   = grow & 1023;
          size_t idx = (((size_t)(b * kH + h) * kN) + n) * kHd + d;
          if (s == 0)      qb[idx] = (_Float16)(val * scale);
          else if (s == 1) kb[idx] = (_Float16)val;
          else             vb[idx] = (_Float16)val;
        } else {
          fout[(size_t)grow * kD + gcol] = val;
        }
      }
    }
  }
}

// ---------------------------------------------------------------------------
// Flash attention: one workgroup per (b, h, 128-query block).
// 8 waves; wave w owns query rows w*16..w*16+15. Streams 64-row K/V tiles,
// double-buffered with async LDS DMA for K; V is transposed during staging.
// ---------------------------------------------------------------------------
__global__ void flash_attn_wmma(const _Float16* __restrict__ q,
                                const _Float16* __restrict__ k,
                                const _Float16* __restrict__ v,
                                _Float16* __restrict__ aout) {
  __shared__ __align__(16) _Float16 Qs[128 * 64];     // [qrow][d]
  __shared__ __align__(16) _Float16 Ks[2][64 * 64];   // [kvrow][d]  (B src for QK^T)
  __shared__ __align__(16) _Float16 Vt[2][64 * 64];   // [d][kvrow]  (B src for PV)
  __shared__ __align__(16) _Float16 Ps[128 * 64];     // per-wave P strip [qrow][kvcol]

  const int qblk = blockIdx.x;           // 0..7
  const int bh   = blockIdx.y;           // 0..191
  const int b    = bh / kH;
  const int h    = bh % kH;
  const int tid  = threadIdx.x;
  const int w    = tid >> 5;
  const int lane = tid & 31;

  const _Float16* qg = q + ((size_t)bh * kN + qblk * 128) * kHd;
  const _Float16* kg = k + (size_t)bh * kN * kHd;
  const _Float16* vg = v + (size_t)bh * kN * kHd;

  // Stage Q block [128][64] = 1024 x b128.
  for (int i = tid; i < 1024; i += 256)
    copy_b128_to_lds(qg + (size_t)i * 8, Qs + (size_t)i * 8);

  auto stage_kv = [&](int buf, int j0) {
    // K tile [64][64]: 512 x b128, straight copy.
    for (int i = tid; i < 512; i += 256)
      copy_b128_to_lds(kg + (size_t)j0 * kHd + (size_t)i * 8, &Ks[buf][i * 8]);
    // V tile transposed: Vt[d][j] = V[j0+j][d] (contiguous global reads,
    // element ds stores; a b128 DMA cannot scatter-transpose).
    int j  = tid >> 2;          // 0..63
    int dp = (tid & 3) * 16;    // 0,16,32,48
    const _Float16* src = vg + (size_t)(j0 + j) * kHd + dp;
#pragma unroll
    for (int i = 0; i < 16; ++i) Vt[buf][(dp + i) * 64 + j] = src[i];
  };

  stage_kv(0, 0);  // prologue (Q + first K/V tile in flight together)

  v8f zero = {};
  v8f o[4];
#pragma unroll
  for (int t = 0; t < 4; ++t) o[t] = zero;
  float mrow[8], lrow[8];
#pragma unroll
  for (int e = 0; e < 8; ++e) { mrow[e] = -1e30f; lrow[e] = 0.0f; }

  const int nl = lane & 15;
  const int mb = (lane < 16) ? 0 : 8;

  for (int jt = 0; jt < kN / 64; ++jt) {
    const int cur = jt & 1;
    wait_async_lds();
    __syncthreads();   // buf[cur] (and Qs on jt==0) now visible to all waves
    if (jt + 1 < kN / 64) stage_kv(cur ^ 1, (jt + 1) * 64);  // overlap

    // S = Q * K^T : 16x64 strip = 4 C tiles, Hd=64 -> 2 WMMA K-steps.
    v8f s[4];
#pragma unroll
    for (int t = 0; t < 4; ++t) s[t] = zero;
#pragma unroll
    for (int kk = 0; kk < 64; kk += 32) {
      v16h a = load_frag_a(Qs + (w * 16) * 64 + kk, 64);
#pragma unroll
      for (int ni = 0; ni < 4; ++ni) {
        v16h bfr = load_frag_b(&Ks[cur][(ni * 16) * 64 + kk], 64);
        s[ni] = WMMA_F16F32(a, bfr, s[ni]);
      }
    }

    // Online softmax. Row M=mb+e lives in one 16-lane half; xor masks 1..8
    // reduce within that half only.
#pragma unroll
    for (int e = 0; e < 8; ++e) {
      float mx = fmaxf(fmaxf(s[0][e], s[1][e]), fmaxf(s[2][e], s[3][e]));
#pragma unroll
      for (int off = 8; off >= 1; off >>= 1) mx = fmaxf(mx, __shfl_xor(mx, off, 32));
      float mn   = fmaxf(mrow[e], mx);
      float corr = __expf(mrow[e] - mn);
      mrow[e] = mn;
      float rs = 0.0f;
#pragma unroll
      for (int t = 0; t < 4; ++t) {
        float ev = __expf(s[t][e] - mn);
        s[t][e] = ev;
        rs += ev;
      }
#pragma unroll
      for (int off = 8; off >= 1; off >>= 1) rs += __shfl_xor(rs, off, 32);
      lrow[e] = lrow[e] * corr + rs;
      o[0][e] *= corr; o[1][e] *= corr; o[2][e] *= corr; o[3][e] *= corr;
    }

    // Re-shape P from C-layout registers to A-layout via wave-private LDS strip.
    {
      _Float16* pw = Ps + (w * 16) * 64;
#pragma unroll
      for (int t = 0; t < 4; ++t)
#pragma unroll
        for (int e = 0; e < 8; ++e)
          pw[(mb + e) * 64 + t * 16 + nl] = (_Float16)s[t][e];
    }
    // Same-wave LDS ops stay in order -> no barrier needed (wave-private strip).

    // O += P * V : K dim = 64 kv rows -> 2 WMMA K-steps.
#pragma unroll
    for (int kk = 0; kk < 64; kk += 32) {
      v16h a = load_frag_a(Ps + (w * 16) * 64 + kk, 64);
#pragma unroll
      for (int ni = 0; ni < 4; ++ni) {
        v16h bfr = load_frag_b(&Vt[cur][(ni * 16) * 64 + kk], 64);
        o[ni] = WMMA_F16F32(a, bfr, o[ni]);
      }
    }
  }

  // Normalize and write to attn-out [B*N][768] (col = h*64 + d), f16.
#pragma unroll
  for (int e = 0; e < 8; ++e) {
    float inv = 1.0f / lrow[e];
    int grow = b * kN + qblk * 128 + w * 16 + mb + e;
#pragma unroll
    for (int t = 0; t < 4; ++t)
      aout[(size_t)grow * kD + h * kHd + t * 16 + nl] = (_Float16)(o[t][e] * inv);
  }
}

// ---------------------------------------------------------------------------
// Launch
// ---------------------------------------------------------------------------
extern "C" void kernel_launch(void* const* d_in, const int* in_sizes, int n_in,
                              void* d_out, int out_size, void* d_ws, size_t ws_size,
                              hipStream_t stream) {
  (void)in_sizes; (void)n_in; (void)out_size; (void)ws_size;

  const float* x      = (const float*)d_in[0];  // [16,1024,768]
  const float* w_qkv  = (const float*)d_in[1];  // [768,2304]
  const float* b_qkv  = (const float*)d_in[2];  // [2304]
  const float* w_proj = (const float*)d_in[3];  // [768,768]
  const float* b_proj = (const float*)d_in[4];  // [768]
  float* out = (float*)d_out;                   // [16,1024,768]

  // Workspace layout (all f16).
  _Float16* p = (_Float16*)d_ws;
  _Float16* xh      = p; p += (size_t)kM * kD;       // x in f16
  _Float16* wqkvT   = p; p += (size_t)kNQKV * kD;    // w_qkv^T
  _Float16* wprojT  = p; p += (size_t)kD * kD;       // w_proj^T
  _Float16* qbuf    = p; p += (size_t)kM * kD;       // [B,H,N,Hd], pre-scaled
  _Float16* kbuf    = p; p += (size_t)kM * kD;
  _Float16* vbuf    = p; p += (size_t)kM * kD;
  _Float16* attnout = p; p += (size_t)kM * kD;       // [B*N,768]

  const float scale = 0.125f;  // 64^-0.5

  // 0) precision conversion / weight transposes
  cvt_f32_to_f16<<<8192, 256, 0, stream>>>(x, xh, (size_t)kM * kD);
  cvt_transpose_f32_to_f16<<<4096, 256, 0, stream>>>(w_qkv, wqkvT, kD, kNQKV);
  cvt_transpose_f32_to_f16<<<2048, 256, 0, stream>>>(w_proj, wprojT, kD, kD);

  // 1) QKV GEMM + bias + scatter (grid: 2304/128 x 16384/128)
  gemm_wmma<0><<<dim3(kNQKV / 128, kM / 128), 256, 0, stream>>>(
      xh, wqkvT, b_qkv, qbuf, kbuf, vbuf, scale, nullptr);

  // 2) flash attention (grid: 8 q-blocks x 192 (b,h) pairs)
  flash_attn_wmma<<<dim3(kN / 128, kB * kH), 256, 0, stream>>>(
      qbuf, kbuf, vbuf, attnout);

  // 3) output projection + bias -> f32 out (grid: 768/128 x 16384/128)
  gemm_wmma<1><<<dim3(kD / 128, kM / 128), 256, 0, stream>>>(
      attnout, wprojT, b_proj, nullptr, nullptr, nullptr, 1.0f, out);
}
